// PointNet2PartSegSsg_21182778704772
// MI455X (gfx1250) — matrix operand order repaired and checked by reading, not compiled, not run on hardware
//
#include <hip/hip_runtime.h>

// ---------------------------------------------------------------------------
// PointNet++ SSG part segmentation forward, CDNA5 (gfx1250), wave32 + WMMA.
// ---------------------------------------------------------------------------

typedef _Float16 f16;
typedef __attribute__((ext_vector_type(16))) _Float16 v16h;
typedef __attribute__((ext_vector_type(4)))  _Float16 v4h;
typedef __attribute__((ext_vector_type(8)))  float    v8f;
typedef __attribute__((ext_vector_type(4)))  int      v4i;

#define NB    16
#define NPTS  4096
#define NCLS  50

// Async global->LDS staging (CDNA5 GLOBAL_LOAD_ASYNC_TO_LDS_B128 + ASYNCcnt),
// compile-safe: falls back to cvt + ds_store staging if builtins are absent.
#if defined(__has_builtin)
#if __has_builtin(__builtin_amdgcn_global_load_async_to_lds_b128) && \
    __has_builtin(__builtin_amdgcn_s_wait_asynccnt)
#define USE_ASYNC_LDS 1
#endif
#endif
#ifndef USE_ASYNC_LDS
#define USE_ASYNC_LDS 0
#endif

#if USE_ASYNC_LDS
typedef __attribute__((address_space(1))) v4i* gv4p;   // global int4*
typedef __attribute__((address_space(3))) v4i* lv4p;   // LDS int4* (32-bit)
#endif

// ---------------------------------------------------------------------------
// Elementwise utility kernels
// ---------------------------------------------------------------------------
__global__ void k_zero(float* p, long n) {
  long i = (long)blockIdx.x * blockDim.x + threadIdx.x;
  if (i < n) p[i] = 0.0f;
}

__global__ void k_copy(const float* __restrict__ s, float* __restrict__ d, long n) {
  long i = (long)blockIdx.x * blockDim.x + threadIdx.x;
  if (i < n) d[i] = s[i];
}

// coord (B,6,N) -> feat0 (B,N,6)
__global__ void k_transpose(const float* __restrict__ c, float* __restrict__ f) {
  long i = (long)blockIdx.x * blockDim.x + threadIdx.x;
  if (i >= (long)NB * NPTS * 6) return;
  int b = (int)(i / (NPTS * 6));
  int r = (int)(i % (NPTS * 6));
  int p = r / 6, ch = r % 6;
  f[i] = c[((long)b * 6 + ch) * NPTS + p];
}

// ---------------------------------------------------------------------------
// Farthest point sampling: one workgroup per batch, dist[] resident in LDS.
// Matches jnp scan semantics: idx[t] = far at entry of step t, far0 = 0,
// argmax tie-break = lowest index.
// ---------------------------------------------------------------------------
__global__ void k_fps(const float* __restrict__ pts, int n, int ldp,
                      int nsample, int* __restrict__ outIdx) {
  __shared__ float dist[4096];
  __shared__ float rv[256];
  __shared__ int   ri[256];
  __shared__ int   sFar;
  int b = blockIdx.x, tid = threadIdx.x;
  const float* base = pts + (size_t)b * n * ldp;
  for (int i = tid; i < n; i += 256) dist[i] = 1e10f;
  if (tid == 0) sFar = 0;
  __syncthreads();
  for (int t = 0; t < nsample; ++t) {
    int far = sFar;
    if (tid == 0) outIdx[b * nsample + t] = far;
    float fx = base[(size_t)far * ldp + 0];
    float fy = base[(size_t)far * ldp + 1];
    float fz = base[(size_t)far * ldp + 2];
    float bv = -1.0f; int bi = 0;
    for (int i = tid; i < n; i += 256) {
      float dx = base[(size_t)i * ldp + 0] - fx;
      float dy = base[(size_t)i * ldp + 1] - fy;
      float dz = base[(size_t)i * ldp + 2] - fz;
      float d  = dx * dx + dy * dy + dz * dz;
      float nd = fminf(dist[i], d);
      dist[i] = nd;
      if (nd > bv) { bv = nd; bi = i; }
    }
    rv[tid] = bv; ri[tid] = bi;
    __syncthreads();
    for (int off = 128; off; off >>= 1) {
      if (tid < off) {
        float ov = rv[tid + off]; int oi = ri[tid + off];
        if (ov > rv[tid] || (ov == rv[tid] && oi < ri[tid])) { rv[tid] = ov; ri[tid] = oi; }
      }
      __syncthreads();
    }
    if (tid == 0) sFar = ri[0];
    __syncthreads();
  }
}

// gather 3 coords per sampled index
__global__ void k_gatherxyz(const float* __restrict__ pts, int n, int ldp,
                            const int* __restrict__ idx, int s, float* __restrict__ o) {
  int i = blockIdx.x * blockDim.x + threadIdx.x;
  if (i >= NB * s) return;
  int b = i / s;
  const float* src = pts + ((size_t)b * n + idx[i]) * ldp;
  o[(size_t)i * 3 + 0] = src[0];
  o[(size_t)i * 3 + 1] = src[1];
  o[(size_t)i * 3 + 2] = src[2];
}

// ---------------------------------------------------------------------------
// Ball query: one wave32 per query point; ascending first-k selection via
// ballot + prefix popcount (== reference's sort-then-take-k), pad with first.
// ---------------------------------------------------------------------------
__global__ void k_ballq(const float* __restrict__ pts, int n, int ldp,
                        const float* __restrict__ nx, int s,
                        float r2, int k, int* __restrict__ gi) {
  int gw   = (blockIdx.x * blockDim.x + threadIdx.x) >> 5;
  int lane = threadIdx.x & 31;
  if (gw >= NB * s) return;
  int b = gw / s;
  const float* qp = nx + (size_t)gw * 3;
  float qx = qp[0], qy = qp[1], qz = qp[2];
  float q2 = qx * qx + qy * qy + qz * qz;
  const float* base = pts + (size_t)b * n * ldp;
  int* out = gi + (size_t)gw * k;
  int cnt = 0, first = -1;
  for (int c0 = 0; c0 < n && cnt < k; c0 += 32) {
    int i = c0 + lane;
    bool in = false;
    if (i < n) {
      float bx = base[(size_t)i * ldp + 0];
      float by = base[(size_t)i * ldp + 1];
      float bz = base[(size_t)i * ldp + 2];
      float d = q2 + bx * bx + by * by + bz * bz - 2.f * (qx * bx + qy * by + qz * bz);
      in = !(d > r2);
    }
    unsigned mm = (unsigned)__ballot(in);
    if (first < 0 && mm) first = c0 + __ffs((int)mm) - 1;
    unsigned below = (lane == 0) ? 0u : (mm & ((1u << lane) - 1u));
    int pos = cnt + __popc(below);
    if (in && pos < k) out[pos] = i;
    cnt += __popc(mm);
  }
  for (int sl = (cnt < k ? cnt : k) + lane; sl < k; sl += 32) out[sl] = first;
}

// ---------------------------------------------------------------------------
// GEMM input (A matrix) builders — write the FULL padded row (zeros in pad).
// ---------------------------------------------------------------------------
__global__ void k_concat_sa1(const float* __restrict__ feat0, const float* __restrict__ x1,
                             const int* __restrict__ gi, float* __restrict__ A) {
  long i = (long)blockIdx.x * blockDim.x + threadIdx.x;       // rows 262144 x 32
  if (i >= (long)NB * 512 * 32 * 32) return;
  long row = i >> 5; int col = (int)(i & 31);
  int b = (int)(row / (512 * 32));
  int j = (int)((row / 32) % 512);
  int g = gi[row];
  float v = 0.0f;
  if (col < 3)      v = feat0[((size_t)b * NPTS + g) * 6 + col] - x1[((size_t)b * 512 + j) * 3 + col];
  else if (col < 9) v = feat0[((size_t)b * NPTS + g) * 6 + (col - 3)];
  A[i] = v;
}

__global__ void k_concat_sa2(const float* __restrict__ x1, const float* __restrict__ x2,
                             const float* __restrict__ f1, const int* __restrict__ gi,
                             float* __restrict__ A) {
  long i = (long)blockIdx.x * blockDim.x + threadIdx.x;       // rows 131072 x 160
  if (i >= (long)NB * 128 * 64 * 160) return;
  long row = i / 160; int col = (int)(i % 160);
  int b = (int)(row / (128 * 64));
  int j = (int)((row / 64) % 128);
  int g = gi[row];
  float v = 0.0f;
  if (col < 3)        v = x1[((size_t)b * 512 + g) * 3 + col] - x2[((size_t)b * 128 + j) * 3 + col];
  else if (col < 131) v = f1[((size_t)b * 512 + g) * 128 + (col - 3)];
  A[i] = v;
}

__global__ void k_concat_sa3(const float* __restrict__ x2, const float* __restrict__ f2,
                             float* __restrict__ A) {
  long i = (long)blockIdx.x * blockDim.x + threadIdx.x;       // rows 2048 x 288
  if (i >= (long)NB * 128 * 288) return;
  long row = i / 288; int col = (int)(i % 288);
  float v = 0.0f;
  if (col < 3)        v = x2[row * 3 + col];
  else if (col < 259) v = f2[row * 256 + (col - 3)];
  A[i] = v;
}

__global__ void k_concat_fp3(const float* __restrict__ f2, const float* __restrict__ f3,
                             float* __restrict__ A) {
  long i = (long)blockIdx.x * blockDim.x + threadIdx.x;       // rows 2048 x 1280
  if (i >= (long)NB * 128 * 1280) return;
  long row = i / 1280; int col = (int)(i % 1280);
  int b = (int)(row >> 7);
  A[i] = (col < 256) ? f2[row * 256 + col] : f3[(size_t)b * 1024 + (col - 256)];
}

__global__ void k_concat_fp2(const float* __restrict__ f1, const float* __restrict__ f2u,
                             const int* __restrict__ ki, const float* __restrict__ kw,
                             float* __restrict__ A) {
  long i = (long)blockIdx.x * blockDim.x + threadIdx.x;       // rows 8192 x 384
  if (i >= (long)NB * 512 * 384) return;
  long row = i / 384; int col = (int)(i % 384);
  int b = (int)(row >> 9);
  if (col < 128) { A[i] = f1[row * 128 + col]; return; }
  int c = col - 128;
  float s = 0.f;
  #pragma unroll
  for (int m = 0; m < 3; ++m)
    s += kw[row * 3 + m] * f2u[((size_t)b * 128 + ki[row * 3 + m]) * 256 + c];
  A[i] = s;
}

__global__ void k_concat_fp1(const float* __restrict__ lab, const float* __restrict__ feat0,
                             const float* __restrict__ f1u,
                             const int* __restrict__ ki, const float* __restrict__ kw,
                             float* __restrict__ A) {
  long i = (long)blockIdx.x * blockDim.x + threadIdx.x;       // rows 65536 x 160
  if (i >= (long)NB * NPTS * 160) return;
  long row = i / 160; int col = (int)(i % 160);
  int b = (int)(row >> 12);
  float v = 0.0f;
  if (col < 16)       v = lab[b * 16 + col];
  else if (col < 19)  v = feat0[row * 6 + (col - 16)];
  else if (col < 25)  v = feat0[row * 6 + (col - 19)];
  else if (col < 153) {
    int c = col - 25; float s = 0.f;
    #pragma unroll
    for (int m = 0; m < 3; ++m)
      s += kw[row * 3 + m] * f1u[((size_t)b * 512 + ki[row * 3 + m]) * 128 + c];
    v = s;
  }
  A[i] = v;
}

// 3-NN with lax.top_k tie semantics (stable, strict-< insertion); weights 1/(d+1e-8)
__global__ void k_knn3(const float* __restrict__ dst, int nd, int ldd,
                       const float* __restrict__ src, int ns, int lds_,
                       int* __restrict__ oid, float* __restrict__ ow) {
  int i = blockIdx.x * blockDim.x + threadIdx.x;
  if (i >= NB * nd) return;
  int b = i / nd;
  const float* a = dst + ((size_t)b * nd + (i % nd)) * ldd;
  float ax = a[0], ay = a[1], az = a[2];
  float a2 = ax * ax + ay * ay + az * az;
  const float* sb = src + (size_t)b * ns * lds_;
  float d0 = 1e30f, d1 = 1e30f, d2 = 1e30f;
  int   i0 = 0, i1 = 0, i2 = 0;
  for (int p = 0; p < ns; ++p) {
    float bx = sb[(size_t)p * lds_ + 0], by = sb[(size_t)p * lds_ + 1], bz = sb[(size_t)p * lds_ + 2];
    float d = a2 + bx * bx + by * by + bz * bz - 2.f * (ax * bx + ay * by + az * bz);
    if (d < d0)      { d2 = d1; i2 = i1; d1 = d0; i1 = i0; d0 = d; i0 = p; }
    else if (d < d1) { d2 = d1; i2 = i1; d1 = d; i1 = p; }
    else if (d < d2) { d2 = d; i2 = p; }
  }
  float w0 = 1.f / (d0 + 1e-8f), w1 = 1.f / (d1 + 1e-8f), w2 = 1.f / (d2 + 1e-8f);
  float s = w0 + w1 + w2;
  oid[i * 3 + 0] = i0; oid[i * 3 + 1] = i1; oid[i * 3 + 2] = i2;
  ow[i * 3 + 0] = w0 / s; ow[i * 3 + 1] = w1 / s; ow[i * 3 + 2] = w2 / s;
}

// ---------------------------------------------------------------------------
// Weight packing: W (N x K, f32) -> B fragments (f16), layout per CDNA5 ISA
// 16-bit B 32x16: lanes 0-15 hold cols n=lane with K {0..7,16..23},
// lanes 16-31 hold K {8..15,24..31}. Zero-pad ragged K/N.
// ---------------------------------------------------------------------------
__global__ void k_pack(const float* __restrict__ W, f16* __restrict__ Bp,
                       int N, int K, int K32, int total) {
  int idx = blockIdx.x * blockDim.x + threadIdx.x;
  if (idx >= total) return;
  int e  = idx & 15;
  int l  = (idx >> 4) & 31;
  int rs = idx >> 9;
  int ks = rs % K32;
  int nt = rs / K32;
  int n  = nt * 16 + (l & 15);
  int k  = (ks << 5) + ((e >> 3) << 4) + ((l >> 4) << 3) + (e & 7);
  float v = (n < N && k < K) ? W[(size_t)n * K + k] : 0.0f;
  Bp[idx] = (f16)v;
}

// ---------------------------------------------------------------------------
// WMMA GEMM: C[M,NC] = A[M,Kp] * W^T ; 256 threads = 8 waves; block tile 32x64.
// A staged f32->f16 through LDS in the ISA A-fragment layout
// (lanes 0-15: K {0..7,16..23}; lanes 16-31: K {8..15,24..31}).
// With USE_ASYNC_LDS, the raw f32 A slab is moved by the async engine
// (GLOBAL_LOAD_ASYNC_TO_LDS_B128 / ASYNCcnt) and converted on the LDS read.
// ---------------------------------------------------------------------------
__global__ __launch_bounds__(256)
void k_gemm(const float* __restrict__ A, const f16* __restrict__ Bp,
            float* __restrict__ C, int M, int Kp, int NC) {
  int tid = threadIdx.x;
  int wave = tid >> 5, lane = tid & 31;
  int wm = wave >> 2, wn = wave & 3;
  int M0    = blockIdx.y * 32;
  int Nbase = blockIdx.x * 64;
  int ntile = (Nbase >> 4) + wn;
  int K32   = Kp >> 5;
  v8f acc = {};

#if USE_ASYNC_LDS
  __shared__ alignas(16) float Araw[32][32];
  // each thread async-copies one 16B chunk of the 4KB f32 slab per k-step
  int trow = tid >> 3;
  int tcol = (tid & 7) << 2;
  const float* gsrc = A + (size_t)(M0 + trow) * Kp + tcol;
  unsigned ldsOff = (unsigned)(size_t)&Araw[trow][tcol];   // low 32 bits = LDS offset
  int r  = wm * 16 + (lane & 15);
  int k0 = (lane & 16) ? 8 : 0;
  for (int ks = 0; ks < K32; ++ks) {
    __syncthreads();   // previous slab fully consumed
    __builtin_amdgcn_global_load_async_to_lds_b128(
        (gv4p)(size_t)(gsrc + (ks << 5)), (lv4p)ldsOff, 0, 0);
    if (ks + 1 < K32)
      __builtin_prefetch(&Bp[(((size_t)ntile * K32 + ks + 1) * 32 + lane) * 16], 0, 1);
    __builtin_amdgcn_s_wait_asynccnt(0);   // this wave's copies landed
    __syncthreads();                       // all waves' copies visible
    v16h a;
    #pragma unroll
    for (int e = 0; e < 8; ++e) a[e]     = (f16)Araw[r][k0 + e];
    #pragma unroll
    for (int e = 0; e < 8; ++e) a[8 + e] = (f16)Araw[r][k0 + 16 + e];
    v16h b = *(const v16h*)(&Bp[(((size_t)ntile * K32 + ks) * 32 + lane) * 16]);
    acc = __builtin_amdgcn_wmma_f32_16x16x32_f16(false, a, false, b,
                                                 (short)0, acc, false, false);
  }
#else
  __shared__ alignas(32) f16 As[2][32][16];
  // per-thread A staging coordinates (one float4 -> 4 halves per k-step)
  int lin = tid << 2;
  int e0  = lin & 15;
  int al  = (lin >> 4) & 31;
  int amt = lin >> 9;
  int am  = al & 15;
  int kb  = ((e0 < 8) ? 0 : 16) + ((al >> 4) << 3) + (e0 & 7);
  const float* Arow = A + (size_t)(M0 + amt * 16 + am) * Kp;
  for (int ks = 0; ks < K32; ++ks) {
    __syncthreads();
    float4 fv = *(const float4*)(Arow + (ks << 5) + kb);
    v4h hv = { (f16)fv.x, (f16)fv.y, (f16)fv.z, (f16)fv.w };
    *(v4h*)&As[amt][al][e0] = hv;
    if (ks + 1 < K32)
      __builtin_prefetch(Arow + ((ks + 1) << 5) + kb, 0, 1);   // global_prefetch_b8
    __syncthreads();
    v16h a = *(const v16h*)(&As[wm][lane][0]);
    v16h b = *(const v16h*)(&Bp[(((size_t)ntile * K32 + ks) * 32 + lane) * 16]);
    acc = __builtin_amdgcn_wmma_f32_16x16x32_f16(false, a, false, b,
                                                 (short)0, acc, false, false);
  }
#endif

  int col   = Nbase + wn * 16 + (lane & 15);
  int rbase = M0 + wm * 16 + ((lane < 16) ? 0 : 8);
  #pragma unroll
  for (int rr = 0; rr < 8; ++rr)
    C[(size_t)(rbase + rr) * NC + col] = acc[rr];
}

// per-channel sum/sumsq over all M rows (global batch statistics)
__global__ void k_stats(const float* __restrict__ C, int M, int NC, float* __restrict__ st) {
  __shared__ float sh[256], sq[256];
  int n = blockIdx.x, tid = threadIdx.x;
  int r0 = blockIdx.y * 8192;
  int r1 = r0 + 8192; if (r1 > M) r1 = M;
  float s = 0.f, q = 0.f;
  for (int r = r0 + tid; r < r1; r += 256) {
    float v = C[(size_t)r * NC + n];
    s += v; q += v * v;
  }
  sh[tid] = s; sq[tid] = q;
  __syncthreads();
  for (int off = 128; off; off >>= 1) {
    if (tid < off) { sh[tid] += sh[tid + off]; sq[tid] += sq[tid + off]; }
    __syncthreads();
  }
  if (tid == 0) { atomicAdd(&st[n], sh[0]); atomicAdd(&st[1024 + n], sq[0]); }
}

__global__ void k_norm(float* __restrict__ C, long tot, int NC, int M,
                       const float* __restrict__ st,
                       const float* __restrict__ g, const float* __restrict__ bb) {
  long i = (long)blockIdx.x * blockDim.x + threadIdx.x;
  if (i >= tot) return;
  int n = (int)(i % NC);
  float mean = st[n] / (float)M;
  float var  = st[1024 + n] / (float)M - mean * mean;
  float v = (C[i] - mean) * rsqrtf(var + 1e-5f) * g[n] + bb[n];
  C[i] = v > 0.f ? v : 0.f;
}

// max over group dim: out[row,c] = max_k in[(row*k+kk), c]
__global__ void k_maxpool(const float* __restrict__ in, float* __restrict__ out,
                          int rows, int k, int C) {
  long i = (long)blockIdx.x * blockDim.x + threadIdx.x;
  if (i >= (long)rows * C) return;
  long row = i / C; int c = (int)(i % C);
  const float* p = in + (size_t)row * k * C + c;
  float m = p[0];
  for (int kk = 1; kk < k; ++kk) m = fmaxf(m, p[(size_t)kk * C]);
  out[i] = m;
}

// logits (rows x 64, cols 0..49 valid) + bias -> log_softmax -> out (rows x 50)
__global__ void k_logsoftmax(const float* __restrict__ L, const float* __restrict__ bias,
                             float* __restrict__ out, int rows) {
  int r = blockIdx.x * blockDim.x + threadIdx.x;
  if (r >= rows) return;
  const float* p = L + (size_t)r * 64;
  float mx = -1e30f;
  for (int c = 0; c < NCLS; ++c) mx = fmaxf(mx, p[c] + bias[c]);
  float s = 0.f;
  for (int c = 0; c < NCLS; ++c) s += expf(p[c] + bias[c] - mx);
  float lg = logf(s);
  for (int c = 0; c < NCLS; ++c) out[(size_t)r * NCLS + c] = p[c] + bias[c] - mx - lg;
}

// ---------------------------------------------------------------------------
// Host orchestration
// ---------------------------------------------------------------------------
struct LayerDesc { int wi, gi, bi, K, Kp, N, NC; };

extern "C" void kernel_launch(void* const* d_in, const int* in_sizes, int n_in,
                              void* d_out, int out_size, void* d_ws, size_t ws_size,
                              hipStream_t stream) {
  (void)in_sizes; (void)n_in; (void)out_size; (void)ws_size;
  const float* coord = (const float*)d_in[0];
  const float* label = (const float*)d_in[1];
  float* out = (float*)d_out;

  float* base = (float*)d_ws;
  size_t off = 0;
  auto alloc = [&](size_t n) { float* p = base + off; off += (n + 7) & ~(size_t)7; return p; };
  float* P0    = alloc(33554432);
  float* P1    = alloc(33554432);
  float* feat0 = alloc((size_t)NB * NPTS * 6);
  float* f1    = alloc((size_t)NB * 512 * 128);
  float* x1    = alloc((size_t)NB * 512 * 3);
  float* f2    = alloc((size_t)NB * 128 * 256);
  float* x2    = alloc((size_t)NB * 128 * 3);
  float* f3    = alloc((size_t)NB * 1024);
  float* f2u   = alloc((size_t)NB * 128 * 256);
  float* f1u   = alloc((size_t)NB * 512 * 128);
  float* stats = alloc(2048);
  int* idx1  = (int*)alloc(NB * 512);
  int* idx2  = (int*)alloc(NB * 128);
  int* gi1   = (int*)alloc((size_t)NB * 512 * 32);
  int* gi2   = (int*)alloc((size_t)NB * 128 * 64);
  int*   knn2i = (int*)alloc((size_t)NB * 512 * 3);
  float* knn2w = alloc((size_t)NB * 512 * 3);
  int*   knn1i = (int*)alloc((size_t)NB * NPTS * 3);
  float* knn1w = alloc((size_t)NB * NPTS * 3);
  f16* wpk = (f16*)alloc(720000);   // 1.44M halves arena

  static const LayerDesc Ls[18] = {
    { 2, 3, 4,    9,  32,  64,  64}, { 5, 6, 7,   64,  64,  64,  64}, { 8, 9,10,   64,  64, 128, 128},
    {11,12,13,  131, 160, 128, 128}, {14,15,16,  128, 128, 128, 128}, {17,18,19,  128, 128, 256, 256},
    {20,21,22,  259, 288, 256, 256}, {23,24,25,  256, 256, 512, 512}, {26,27,28,  512, 512,1024,1024},
    {29,30,31, 1280,1280, 256, 256}, {32,33,34,  256, 256, 256, 256},
    {35,36,37,  384, 384, 256, 256}, {38,39,40,  256, 256, 128, 128},
    {41,42,43,  153, 160, 128, 128}, {44,45,46,  128, 128, 128, 128}, {47,48,49,  128, 128, 128, 128},
    {50,51,52,  128, 128, 128, 128},
    {53,-1,-1,  128, 128,  50,  64},
  };
  size_t bp[18]; size_t acc = 0;
  for (int i = 0; i < 18; ++i) {
    bp[i] = acc;
    acc += (size_t)(Ls[i].NC / 16) * (Ls[i].Kp / 32) * 512;
  }
  // pack all weights (deterministic, every call)
  for (int i = 0; i < 18; ++i) {
    int K32 = Ls[i].Kp / 32;
    int tot = (Ls[i].NC / 16) * K32 * 512;
    k_pack<<<(tot + 255) / 256, 256, 0, stream>>>((const float*)d_in[Ls[i].wi],
                                                  wpk + bp[i], Ls[i].N, Ls[i].K, K32, tot);
  }

  auto runL = [&](int li, const float* A, float* Cb, int M, bool ln) {
    const LayerDesc& L = Ls[li];
    dim3 g(L.NC / 64, M / 32);
    k_gemm<<<g, 256, 0, stream>>>(A, wpk + bp[li], Cb, M, L.Kp, L.NC);
    if (ln) {
      k_zero<<<8, 256, 0, stream>>>(stats, 2048);
      dim3 gs(L.NC, (M + 8191) / 8192);
      k_stats<<<gs, 256, 0, stream>>>(Cb, M, L.NC, stats);
      long tot = (long)M * L.NC;
      k_norm<<<(unsigned)((tot + 255) / 256), 256, 0, stream>>>(
          Cb, tot, L.NC, M, stats,
          (const float*)d_in[L.gi], (const float*)d_in[L.bi]);
    }
  };
  auto nblk = [](long n) { return (unsigned)((n + 255) / 256); };

  // ---- stage 0: transpose ----
  k_transpose<<<nblk((long)NB * NPTS * 6), 256, 0, stream>>>(coord, feat0);

  // ---- SA1 ----
  k_fps<<<NB, 256, 0, stream>>>(feat0, NPTS, 6, 512, idx1);
  k_gatherxyz<<<nblk(NB * 512), 256, 0, stream>>>(feat0, NPTS, 6, idx1, 512, x1);
  k_ballq<<<nblk((long)NB * 512 * 32), 256, 0, stream>>>(feat0, NPTS, 6, x1, 512, 0.04f, 32, gi1);
  k_concat_sa1<<<nblk((long)NB * 512 * 32 * 32), 256, 0, stream>>>(feat0, x1, gi1, P0);
  int M1 = NB * 512 * 32;
  runL(0, P0, P1, M1, true);
  runL(1, P1, P0, M1, true);
  runL(2, P0, P1, M1, true);
  k_maxpool<<<nblk((long)NB * 512 * 128), 256, 0, stream>>>(P1, f1, NB * 512, 32, 128);

  // ---- SA2 ----
  k_fps<<<NB, 256, 0, stream>>>(x1, 512, 3, 128, idx2);
  k_gatherxyz<<<nblk(NB * 128), 256, 0, stream>>>(x1, 512, 3, idx2, 128, x2);
  k_ballq<<<nblk((long)NB * 128 * 32), 256, 0, stream>>>(x1, 512, 3, x2, 128, 0.16f, 64, gi2);
  k_concat_sa2<<<nblk((long)NB * 128 * 64 * 160), 256, 0, stream>>>(x1, x2, f1, gi2, P0);
  int M2 = NB * 128 * 64;
  runL(3, P0, P1, M2, true);
  runL(4, P1, P0, M2, true);
  runL(5, P0, P1, M2, true);
  k_maxpool<<<nblk((long)NB * 128 * 256), 256, 0, stream>>>(P1, f2, NB * 128, 64, 256);

  // ---- SA3 (group-all) ----
  k_concat_sa3<<<nblk((long)NB * 128 * 288), 256, 0, stream>>>(x2, f2, P0);
  int M3 = NB * 128;
  runL(6, P0, P1, M3, true);
  runL(7, P1, P0, M3, true);
  runL(8, P0, P1, M3, true);
  k_maxpool<<<nblk((long)NB * 1024), 256, 0, stream>>>(P1, f3, NB, 128, 1024);

  // ---- FP3 ----
  k_concat_fp3<<<nblk((long)NB * 128 * 1280), 256, 0, stream>>>(f2, f3, P0);
  runL(9, P0, P1, M3, true);
  runL(10, P1, P0, M3, true);
  k_copy<<<nblk((long)NB * 128 * 256), 256, 0, stream>>>(P0, f2u, (long)NB * 128 * 256);

  // ---- FP2 ----
  k_knn3<<<nblk(NB * 512), 256, 0, stream>>>(x1, 512, 3, x2, 128, 3, knn2i, knn2w);
  k_concat_fp2<<<nblk((long)NB * 512 * 384), 256, 0, stream>>>(f1, f2u, knn2i, knn2w, P0);
  int M4 = NB * 512;
  runL(11, P0, P1, M4, true);
  runL(12, P1, P0, M4, true);
  k_copy<<<nblk((long)NB * 512 * 128), 256, 0, stream>>>(P0, f1u, (long)NB * 512 * 128);

  // ---- FP1 ----
  k_knn3<<<nblk(NB * NPTS), 256, 0, stream>>>(feat0, NPTS, 6, x1, 512, 3, knn1i, knn1w);
  k_concat_fp1<<<nblk((long)NB * NPTS * 160), 256, 0, stream>>>(label, feat0, f1u, knn1i, knn1w, P0);
  int M5 = NB * NPTS;
  runL(13, P0, P1, M5, true);
  runL(14, P1, P0, M5, true);
  runL(15, P0, P1, M5, true);

  // ---- head ----
  runL(16, P1, P0, M5, true);    // head_mlp -> h in P0
  runL(17, P0, P1, M5, false);   // logits (rows x 64) in P1
  k_logsoftmax<<<nblk(M5), 256, 0, stream>>>(P1, (const float*)d_in[54], out, M5);

  // coord3 = transpose(zeros (B,1,3)) -> 48 zeros appended
  k_zero<<<1, 64, 0, stream>>>(out + (long)NB * NPTS * NCLS, 48);
}
